// differentiableTokenizerVisionTransformer_28132035788971
// MI455X (gfx1250) — matrix operand order repaired and checked by reading, not compiled, not run on hardware
//
#include <hip/hip_runtime.h>
#include <hip/hip_bf16.h>
#include <stdint.h>

// ---------------------------------------------------------------------------
// MI455X (gfx1250): all dense linear algebra via v_wmma_f32_16x16x32_f16
// (f16 in, f32 accumulate). B panels double-buffered into LDS with
// global_load_async_to_lds_b128 (ASYNCcnt path) so DMA overlaps compute.
// Residual stream kept f32.
// ---------------------------------------------------------------------------

typedef _Float16 h16;
typedef _Float16 v8h  __attribute__((ext_vector_type(8)));
typedef _Float16 v16h __attribute__((ext_vector_type(16)));
typedef float    v8f  __attribute__((ext_vector_type(8)));

#define BDIM  2
#define CN    3
#define HIW   224
#define GRIDN 14
#define SSEG  196
#define PPIX  256
#define DD    384
#define DEPTH 12
#define HID   1536
#define NCLS  1000
#define NP    224     // ViT tokens padded 197 -> 224 (multiple of 32)
#define NTOK  197
#define CH    56      // tokenizer chunk: sequences per pass
#define NCHUNK 7      // 392 / 56

// ---------------------------------------------------------------------------
// Batched WMMA GEMM:  C[M,N] = A[M,K] * B[N,K]^T (+bias) (+C  if ACC)
// Both operands row-major with K contiguous (torch x @ W.T / q @ k.T; attn@v
// uses a pre-transposed V so it is the same shape).
//   CF16 : write f16 output (feeds the next WMMA), else f32
//   ACC  : accumulator preloaded from Cf (fused residual add: x += ...)
// Block = 4 waves -> 64x64 C macro-block; each wave owns a 32x32 quadrant
// (4 WMMA tiles, 4 accumulators) for 4 wmma per K-step body.
// The 64(N)x32(K) B panel is double-buffered in LDS: the next panel's
// global_load_async_to_lds_b128 pair is issued before waiting on the current
// one (in-order completion => s_wait_asynccnt 2 retires exactly the current
// panel), so the async DMA overlaps the WMMA work.
// Batch z = z0*Z1 + z1 with independent strides (sequence x head batching).
// ---------------------------------------------------------------------------
template<int CF16, int ACC>
__global__ __launch_bounds__(128)
void gemm16(const h16* __restrict__ A, long long sA0, long long sA1, int lda,
            const h16* __restrict__ B, long long sB0, long long sB1, int ldb,
            const float* __restrict__ bias,
            float* __restrict__ Cf, h16* __restrict__ Ch,
            long long sC0, long long sC1, int ldc,
            int M, int N, int K, int Z1)
{
    __shared__ h16 bt[2][64 * 32];                // double-buffered B panel

    const int z  = blockIdx.z;
    const int z0 = z / Z1, z1 = z % Z1;
    A += (long long)z0 * sA0 + (long long)z1 * sA1;
    B += (long long)z0 * sB0 + (long long)z1 * sB1;
    const long long coff = (long long)z0 * sC0 + (long long)z1 * sC1;

    const int t    = threadIdx.x;
    const int wave = t >> 5;
    const int lane = t & 31;
    const int r    = lane & 15;       // row (A) / col (B,C) within tile
    const int hi   = lane >> 4;       // lane half selector
    const int koff = hi * 16;         // A: K chunk koff..koff+15 (contiguous)

    const int n0blk = blockIdx.x * 64;
    const int nw = n0blk + (wave & 1) * 32;       // wave's N base
    const int mw = blockIdx.y * 64 + (wave >> 1) * 32;   // wave's M base

    bool actm[2], actn[2];
    #pragma unroll
    for (int i = 0; i < 2; ++i) {
        actm[i] = (mw + i * 16) < M;              // dims are multiples of 16
        actn[i] = (nw + i * 16) < N;
    }

    // staging map: thread t copies B row (n0blk + t>>1), 32B at k-seg (t&1)*16
    const int srow = t >> 1;
    const int sk   = (t & 1) * 16;
    const int grow = (n0blk + srow < N) ? (n0blk + srow) : (N - 1);  // clamp
    const h16* gB = B + (long long)grow * ldb + sk;
    const uint32_t lds0 = (uint32_t)(uintptr_t)&bt[0][srow * 32 + sk];
    const uint32_t lds1 = (uint32_t)(uintptr_t)&bt[1][srow * 32 + sk];

    auto stage = [&](int buf, int k0) {           // 2 async ops per wave
        const uint64_t gp = (uint64_t)(uintptr_t)(gB + k0);
        const uint32_t lo = buf ? lds1 : lds0;
        asm volatile("global_load_async_to_lds_b128 %0, %1, off"
                     :: "v"(lo), "v"(gp) : "memory");
        asm volatile("global_load_async_to_lds_b128 %0, %1, off offset:16"
                     :: "v"(lo), "v"(gp) : "memory");
    };

    v8f c[2][2] = {};
    if (ACC) {
        #pragma unroll
        for (int mt = 0; mt < 2; ++mt)
            #pragma unroll
            for (int nt = 0; nt < 2; ++nt)
                if (actm[mt] && actn[nt]) {
                    const float* cp = Cf + coff
                        + (long long)(mw + mt * 16 + hi * 8) * ldc + nw + nt * 16 + r;
                    #pragma unroll
                    for (int i = 0; i < 8; ++i) c[mt][nt][i] = cp[(long long)i * ldc];
                }
    }

    stage(0, 0);                                  // prologue prefetch
    for (int k0 = 0; k0 < K; k0 += 32) {
        const int  cur     = (k0 >> 5) & 1;
        const bool hasnext = (k0 + 32) < K;
        if (hasnext) {                            // prefetch next panel first
            stage(cur ^ 1, k0 + 32);
            asm volatile("s_wait_asynccnt 0x2" ::: "memory");  // cur complete
        } else {
            asm volatile("s_wait_asynccnt 0x0" ::: "memory");
        }
        __syncthreads();

        v16h af[2];
        #pragma unroll
        for (int mt = 0; mt < 2; ++mt)
            if (actm[mt]) {   // A fragment: contiguous K order, element j = k0 + koff + j
                const h16* ap = A + (long long)(mw + mt * 16 + r) * lda + k0 + koff;
                v8h lo = *(const v8h*)ap;
                v8h hh = *(const v8h*)(ap + 8);
                #pragma unroll
                for (int j = 0; j < 8; ++j) { af[mt][j] = lo[j]; af[mt][j + 8] = hh[j]; }
            }
        #pragma unroll
        for (int nt = 0; nt < 2; ++nt) {
            // B fragment: contiguous 32B from LDS (lane=col, K contiguous)
            const h16* bp = &bt[cur][(size_t)((wave & 1) * 32 + nt * 16 + r) * 32 + hi * 16];
            v16h bf = *(const v16h*)bp;
            #pragma unroll
            for (int mt = 0; mt < 2; ++mt)
                if (actm[mt] && actn[nt])
                    c[mt][nt] = __builtin_amdgcn_wmma_f32_16x16x32_f16(
                        false, af[mt], false, bf, (short)0, c[mt][nt], false, false);
        }
        __syncthreads();                          // protect cur from overwrite
    }

    #pragma unroll
    for (int mt = 0; mt < 2; ++mt)
        #pragma unroll
        for (int nt = 0; nt < 2; ++nt)
            if (actm[mt] && actn[nt]) {
                const int col  = nw + nt * 16 + r;
                const int mrow = mw + mt * 16 + hi * 8;
                const float bv = bias ? bias[col] : 0.f;
                #pragma unroll
                for (int i = 0; i < 8; ++i) {
                    const float v = c[mt][nt][i] + bv;
                    const long long off = coff + (long long)(mrow + i) * ldc + col;
                    if (CF16) Ch[off] = (h16)v;
                    else      Cf[off] = v;
                }
            }
}

// --------------------------- elementwise kernels ---------------------------

__global__ void cast_f32_f16(const float* __restrict__ x, h16* __restrict__ y, int n)
{
    int i = blockIdx.x * blockDim.x + threadIdx.x;
    if (i < n) y[i] = (h16)x[i];
}

// superpixel gather (regular 16x16 grid) + Linear(C=3 -> D) -> f16 feats[b,s,p,d]
__global__ void feats_kernel(const float* __restrict__ img,
                             const float* __restrict__ fp_w,
                             const float* __restrict__ fp_b,
                             h16* __restrict__ out)
{
    long long i = (long long)blockIdx.x * blockDim.x + threadIdx.x;
    const long long total = (long long)BDIM * SSEG * PPIX * DD;
    if (i >= total) return;
    int d = (int)(i % DD);
    int p = (int)((i / DD) % PPIX);
    int s = (int)((i / ((long long)DD * PPIX)) % SSEG);
    int b = (int)(i / ((long long)DD * PPIX * SSEG));
    int gy = s / GRIDN, gx = s % GRIDN;
    int py = p >> 4,    px = p & 15;
    int y = gy * 16 + py, x = gx * 16 + px;
    float acc = fp_b[d];
    #pragma unroll
    for (int cc = 0; cc < CN; ++cc)
        acc += img[(((long long)b * CN + cc) * HIW + y) * HIW + x] * fp_w[d * CN + cc];
    out[i] = (h16)acc;
}

// transpose per-head V slices out of the packed qkv buffer:
//   vt[((s*H + h)*dh + e)*NK + j] = qkv[(s*NK + j)*3*DD + 2*DD + h*dh + e]
__global__ void transpose_v(const h16* __restrict__ qkv, h16* __restrict__ vt,
                            int NK, int H, int dh, long long total)
{
    long long i = (long long)blockIdx.x * blockDim.x + threadIdx.x;
    if (i >= total) return;
    int j = (int)(i % NK);
    int e = (int)((i / NK) % dh);
    int h = (int)((i / ((long long)NK * dh)) % H);
    long long s = i / ((long long)NK * dh * H);
    vt[i] = qkv[(s * NK + j) * (3 * DD) + 2 * DD + h * dh + e];
}

// mean over the 256 tokens of each pooled superpixel sequence
__global__ void mean_ctx(const h16* __restrict__ ctx, h16* __restrict__ mean, int c0)
{
    int i = blockIdx.x * blockDim.x + threadIdx.x;
    if (i >= CH * DD) return;
    int d = i % DD, si = i / DD;
    const h16* p = ctx + (long long)si * PPIX * DD + d;
    float s = 0.f;
    for (int pix = 0; pix < PPIX; ++pix) s += (float)p[(long long)pix * DD];
    mean[(long long)(c0 + si) * DD + d] = (h16)(s * (1.f / PPIX));
}

// build padded residual stream x[b, 0..223, d]; rows >=197 zeroed
__global__ void x_init(const float* __restrict__ tok, const float* __restrict__ cls,
                       const float* __restrict__ pos, float* __restrict__ x)
{
    int i = blockIdx.x * blockDim.x + threadIdx.x;
    if (i >= BDIM * NP * DD) return;
    int d = i % DD;
    int n = (i / DD) % NP;
    int b = i / (DD * NP);
    float v = 0.f;
    if (n == 0)          v = cls[d] + pos[d];
    else if (n <= SSEG)  v = tok[(long long)(b * SSEG + n - 1) * DD + d] + pos[n * DD + d];
    x[i] = v;
}

// LayerNorm over D=384; f32 in, f16 out (feeds WMMA)
__global__ __launch_bounds__(384)
void ln_rows(const float* __restrict__ X, const float* __restrict__ g,
             const float* __restrict__ bt, h16* __restrict__ Y)
{
    __shared__ float s1[384], s2[384];
    int row = blockIdx.x, t = threadIdx.x;
    float v = X[(long long)row * DD + t];
    s1[t] = v; s2[t] = v * v;
    __syncthreads();
    if (t < 128) { s1[t] += s1[t + 256]; s2[t] += s2[t + 256]; }
    __syncthreads();
    for (int s = 128; s > 0; s >>= 1) {
        if (t < s) { s1[t] += s1[t + s]; s2[t] += s2[t + s]; }
        __syncthreads();
    }
    float m  = s1[0] * (1.f / DD);
    float va = s2[0] * (1.f / DD) - m * m;
    float rs = rsqrtf(va + 1e-6f);
    Y[(long long)row * DD + t] = (h16)((v - m) * rs * g[t] + bt[t]);
}

// masked, scaled softmax over a row; f32 scores -> f16 attn (keys >= valid -> 0)
__global__ __launch_bounds__(256)
void softmax_rows(const float* __restrict__ S, h16* __restrict__ A,
                  int N, int valid, float scale)
{
    __shared__ float sh[256];
    long long row = blockIdx.x;
    int t = threadIdx.x;
    float v = (t < valid) ? S[row * N + t] * scale : -3.0e38f;
    sh[t] = v; __syncthreads();
    for (int s = 128; s > 0; s >>= 1) { if (t < s) sh[t] = fmaxf(sh[t], sh[t + s]); __syncthreads(); }
    float m = sh[0]; __syncthreads();
    float e = (t < valid) ? __expf(v - m) : 0.f;
    sh[t] = e; __syncthreads();
    for (int s = 128; s > 0; s >>= 1) { if (t < s) sh[t] += sh[t + s]; __syncthreads(); }
    float inv = 1.f / sh[0];
    if (t < N) A[row * N + t] = (h16)(e * inv);
}

// exact GELU (erf) f32 -> f16
__global__ void gelu_f32_f16(const float* __restrict__ x, h16* __restrict__ y, int n)
{
    int i = blockIdx.x * blockDim.x + threadIdx.x;
    if (i < n) {
        float v = x[i];
        y[i] = (h16)(0.5f * v * (1.f + erff(v * 0.70710678118654752f)));
    }
}

// classifier head: out[b,c] = xn[b,0,:] . head_w[c,:] + head_b[c]   (tiny)
__global__ void head_k(const h16* __restrict__ xn, const float* __restrict__ w,
                       const float* __restrict__ b, float* __restrict__ out)
{
    int i = blockIdx.x * blockDim.x + threadIdx.x;
    if (i >= BDIM * NCLS) return;
    int c  = i % NCLS, bb = i / NCLS;
    const h16* xr = xn + (long long)bb * NP * DD;   // cls token row
    float s = b[c];
    for (int d = 0; d < DD; ++d) s += (float)xr[d] * w[(long long)c * DD + d];
    out[i] = s;
}

// ------------------------------ host side ----------------------------------

template<int CF16, int ACC>
static inline void gemm_launch(hipStream_t st,
    const h16* A, long long sA0, long long sA1, int lda,
    const h16* B, long long sB0, long long sB1, int ldb,
    const float* bias, float* Cf, h16* Ch,
    long long sC0, long long sC1, int ldc,
    int M, int N, int K, int Z0, int Z1)
{
    dim3 g((N + 63) / 64, (M + 63) / 64, Z0 * Z1), b(128, 1, 1);
    gemm16<CF16, ACC><<<g, b, 0, st>>>(A, sA0, sA1, lda, B, sB0, sB1, ldb,
                                       bias, Cf, Ch, sC0, sC1, ldc, M, N, K, Z1);
}

extern "C" void kernel_launch(void* const* d_in, const int* in_sizes, int n_in,
                              void* d_out, int out_size, void* d_ws, size_t ws_size,
                              hipStream_t stream)
{
    (void)in_sizes; (void)n_in; (void)out_size; (void)ws_size;

    const float* img    = (const float*)d_in[0];
    const float* fp_w   = (const float*)d_in[2];
    const float* fp_b   = (const float*)d_in[3];
    const float* apin_w = (const float*)d_in[4];
    const float* apin_b = (const float*)d_in[5];
    const float* apout_w= (const float*)d_in[6];
    const float* apout_b= (const float*)d_in[7];
    const float* cls    = (const float*)d_in[8];
    const float* pos    = (const float*)d_in[9];
    const float* ln1_g  = (const float*)d_in[10];
    const float* ln1_b  = (const float*)d_in[11];
    const float* qkv_w  = (const float*)d_in[12];
    const float* qkv_b  = (const float*)d_in[13];
    const float* proj_w = (const float*)d_in[14];
    const float* proj_b = (const float*)d_in[15];
    const float* ln2_g  = (const float*)d_in[16];
    const float* ln2_b  = (const float*)d_in[17];
    const float* fc1_w  = (const float*)d_in[18];
    const float* fc1_b  = (const float*)d_in[19];
    const float* fc2_w  = (const float*)d_in[20];
    const float* fc2_b  = (const float*)d_in[21];
    const float* nrm_g  = (const float*)d_in[22];
    const float* nrm_b  = (const float*)d_in[23];
    const float* head_w = (const float*)d_in[24];
    const float* head_b = (const float*)d_in[25];
    float* out = (float*)d_out;

    // ---- workspace bump allocator (256B aligned) ----
    size_t off = 0;
    auto alloc = [&](size_t bytes) -> char* {
        char* p = (char*)d_ws + off;
        off = (off + bytes + 255) & ~(size_t)255;
        return p;
    };
    h16*   wF     = (h16*)  alloc((size_t)BDIM * SSEG * PPIX * DD * 2);   // 77 MB feats
    h16*   wWqkv  = (h16*)  alloc((size_t)DEPTH * 3 * DD * DD * 2);
    h16*   wWproj = (h16*)  alloc((size_t)DEPTH * DD * DD * 2);
    h16*   wWfc1  = (h16*)  alloc((size_t)DEPTH * HID * DD * 2);
    h16*   wWfc2  = (h16*)  alloc((size_t)DEPTH * DD * HID * 2);
    h16*   wWapin = (h16*)  alloc((size_t)3 * DD * DD * 2);
    h16*   wWapout= (h16*)  alloc((size_t)DD * DD * 2);
    h16*   wQKVt  = (h16*)  alloc((size_t)CH * PPIX * 3 * DD * 2);        // chunk qkv
    h16*   wVtT   = (h16*)  alloc((size_t)CH * 4 * 96 * PPIX * 2);        // chunk V^T
    float* wScT   = (float*)alloc((size_t)CH * 4 * PPIX * PPIX * 4);      // chunk scores
    h16*   wAtT   = (h16*)  alloc((size_t)CH * 4 * PPIX * PPIX * 2);      // chunk attn
    h16*   wCtxT  = (h16*)  alloc((size_t)CH * PPIX * DD * 2);            // chunk ctx
    h16*   wMean  = (h16*)  alloc((size_t)400 * DD * 2);                  // 392 padded to 400
    float* wTok   = (float*)alloc((size_t)400 * DD * 4);
    float* wX     = (float*)alloc((size_t)BDIM * NP * DD * 4);            // residual stream
    h16*   wXn    = (h16*)  alloc((size_t)BDIM * NP * DD * 2);
    h16*   wVqkv  = (h16*)  alloc((size_t)BDIM * NP * 3 * DD * 2);
    h16*   wVtV   = (h16*)  alloc((size_t)BDIM * 6 * 64 * NP * 2);        // ViT V^T
    float* wVsc   = (float*)alloc((size_t)BDIM * 6 * NP * NP * 4);
    h16*   wVat   = (h16*)  alloc((size_t)BDIM * 6 * NP * NP * 2);
    h16*   wVctx  = (h16*)  alloc((size_t)BDIM * NP * DD * 2);
    float* wMlp   = (float*)alloc((size_t)BDIM * NP * HID * 4);
    h16*   wHmlp  = (h16*)  alloc((size_t)BDIM * NP * HID * 2);

    auto cast = [&](const float* s, h16* d, long long n) {
        cast_f32_f16<<<(unsigned)((n + 255) / 256), 256, 0, stream>>>(s, d, (int)n);
    };
    // ---- weights f32 -> f16 ----
    cast(qkv_w,  wWqkv,  (long long)DEPTH * 3 * DD * DD);
    cast(proj_w, wWproj, (long long)DEPTH * DD * DD);
    cast(fc1_w,  wWfc1,  (long long)DEPTH * HID * DD);
    cast(fc2_w,  wWfc2,  (long long)DEPTH * DD * HID);
    cast(apin_w, wWapin, (long long)3 * DD * DD);
    cast(apout_w,wWapout,(long long)DD * DD);

    // ---- tokenizer: patch gather + Linear(3->384) ----
    {
        long long tot = (long long)BDIM * SSEG * PPIX * DD;
        feats_kernel<<<(unsigned)((tot + 255) / 256), 256, 0, stream>>>(img, fp_w, fp_b, wF);
    }

    // ---- tokenizer attention pooling, chunked (56 sequences / pass) ----
    const float tscale = 0.102062072615966f;   // 1/sqrt(96)
    for (int c = 0; c < NCHUNK; ++c) {
        const int c0 = c * CH;
        // qkv = feats @ apin_w.T + b    [CH*256, 1152] f16
        gemm_launch<1,0>(stream, wF + (long long)c0 * PPIX * DD, 0, 0, DD,
                         wWapin, 0, 0, DD, apin_b, nullptr, wQKVt,
                         0, 0, 3 * DD, CH * PPIX, 3 * DD, DD, 1, 1);
        // scores = q @ k.T  per (seq, head) : dh=96
        gemm_launch<0,0>(stream, wQKVt, (long long)PPIX * 3 * DD, 96, 3 * DD,
                         wQKVt + DD, (long long)PPIX * 3 * DD, 96, 3 * DD,
                         nullptr, wScT, nullptr,
                         (long long)4 * PPIX * PPIX, (long long)PPIX * PPIX, PPIX,
                         PPIX, PPIX, 96, CH, 4);
        softmax_rows<<<CH * 4 * PPIX, 256, 0, stream>>>(wScT, wAtT, PPIX, PPIX, tscale);
        // V^T then ctx = attn @ v  per (seq, head)
        {
            long long tot = (long long)CH * 4 * 96 * PPIX;
            transpose_v<<<(unsigned)((tot + 255) / 256), 256, 0, stream>>>(
                wQKVt, wVtT, PPIX, 4, 96, tot);
        }
        gemm_launch<1,0>(stream, wAtT, (long long)4 * PPIX * PPIX, (long long)PPIX * PPIX, PPIX,
                         wVtT, (long long)4 * 96 * PPIX, (long long)96 * PPIX, PPIX,
                         nullptr, nullptr, wCtxT,
                         (long long)PPIX * DD, 96, DD,
                         PPIX, 96, PPIX, CH, 4);
        // mean over the 256 tokens (out-proj commutes with the mean)
        mean_ctx<<<(CH * DD + 255) / 256, 256, 0, stream>>>(wCtxT, wMean, c0);
    }
    // tok = mean(ctx) @ apout_w.T + b    [400(392), 384] f32
    gemm_launch<0,0>(stream, wMean, 0, 0, DD, wWapout, 0, 0, DD,
                     apout_b, wTok, nullptr, 0, 0, DD, 400, DD, DD, 1, 1);

    // ---- assemble padded residual stream ----
    x_init<<<(BDIM * NP * DD + 255) / 256, 256, 0, stream>>>(wTok, cls, pos, wX);

    // ---- ViT trunk: 12 pre-norm blocks ----
    const int ROWS = BDIM * NP;                 // 448
    for (int l = 0; l < DEPTH; ++l) {
        ln_rows<<<ROWS, DD, 0, stream>>>(wX, ln1_g + l * DD, ln1_b + l * DD, wXn);
        gemm_launch<1,0>(stream, wXn, 0, 0, DD,
                         wWqkv + (long long)l * 3 * DD * DD, 0, 0, DD,
                         qkv_b + (long long)l * 3 * DD, nullptr, wVqkv,
                         0, 0, 3 * DD, ROWS, 3 * DD, DD, 1, 1);
        // scores per (batch, head) : dh=64
        gemm_launch<0,0>(stream, wVqkv, (long long)NP * 3 * DD, 64, 3 * DD,
                         wVqkv + DD, (long long)NP * 3 * DD, 64, 3 * DD,
                         nullptr, wVsc, nullptr,
                         (long long)6 * NP * NP, (long long)NP * NP, NP,
                         NP, NP, 64, BDIM, 6);
        softmax_rows<<<BDIM * 6 * NP, 256, 0, stream>>>(wVsc, wVat, NP, NTOK, 0.125f);
        {
            long long tot = (long long)BDIM * 6 * 64 * NP;
            transpose_v<<<(unsigned)((tot + 255) / 256), 256, 0, stream>>>(
                wVqkv, wVtV, NP, 6, 64, tot);
        }
        gemm_launch<1,0>(stream, wVat, (long long)6 * NP * NP, (long long)NP * NP, NP,
                         wVtV, (long long)6 * 64 * NP, (long long)64 * NP, NP,
                         nullptr, nullptr, wVctx,
                         (long long)NP * DD, 64, DD,
                         NP, 64, NP, BDIM, 6);
        // x += ctx @ proj_w.T + proj_b   (fused residual)
        gemm_launch<0,1>(stream, wVctx, 0, 0, DD,
                         wWproj + (long long)l * DD * DD, 0, 0, DD,
                         proj_b + (long long)l * DD, wX, nullptr,
                         0, 0, DD, ROWS, DD, DD, 1, 1);
        // MLP
        ln_rows<<<ROWS, DD, 0, stream>>>(wX, ln2_g + l * DD, ln2_b + l * DD, wXn);
        gemm_launch<0,0>(stream, wXn, 0, 0, DD,
                         wWfc1 + (long long)l * HID * DD, 0, 0, DD,
                         fc1_b + (long long)l * HID, wMlp, nullptr,
                         0, 0, HID, ROWS, HID, DD, 1, 1);
        gelu_f32_f16<<<(ROWS * HID + 255) / 256, 256, 0, stream>>>(wMlp, wHmlp, ROWS * HID);
        gemm_launch<0,1>(stream, wHmlp, 0, 0, HID,
                         wWfc2 + (long long)l * DD * HID, 0, 0, HID,
                         fc2_b + (long long)l * DD, wX, nullptr,
                         0, 0, DD, ROWS, DD, HID, 1, 1);
    }

    // ---- final norm + head ----
    ln_rows<<<ROWS, DD, 0, stream>>>(wX, nrm_g, nrm_b, wXn);
    head_k<<<(BDIM * NCLS + 127) / 128, 128, 0, stream>>>(wXn, head_w, head_b, out);
}
